// _InnerAttention_45638322487505
// MI455X (gfx1250) — matrix operand order repaired and checked by reading, not compiled, not run on hardware
//
#include <hip/hip_runtime.h>

typedef __attribute__((ext_vector_type(16))) _Float16 v16h;
typedef __attribute__((ext_vector_type(8)))  _Float16 v8h;
typedef __attribute__((ext_vector_type(4)))  _Float16 v4h;
typedef __attribute__((ext_vector_type(8)))  float    v8f;
typedef __attribute__((ext_vector_type(4)))  float    v4f;

#define NHEADS 8
#define ZDIM   128
#define VDIM   256
#define CHUNK  1024
#define QT     128           // q rows per workgroup (8 waves x 16)
#define KT     128           // keys per k-tile
#define ROWH   136           // padded LDS row stride in halfs (272B, 16B-aligned, conflict-free)
#define NEGB   (-1.0e9f)

#define QS_OFF 0
#define KS_OFF (QT * ROWH)
#define VT_OFF (KS_OFF + KT * ROWH)
#define SMEM_HALFS (VT_OFF + VDIM * ROWH)
#define SMEM_BYTES (SMEM_HALFS * 2)

union HF  { v16h v; v8h h[2]; };
union PU  { v8h  h; int i[4]; };

__global__ void __launch_bounds__(256)
fa_chunked_rope_kernel(const float* __restrict__ xq, const float* __restrict__ xk,
                       const float* __restrict__ xv, const float* __restrict__ cs,
                       const float* __restrict__ sn, float* __restrict__ out) {
  extern __shared__ _Float16 smem[];
  _Float16* Qs = smem + QS_OFF;   // [QT][ROWH]  q rows (f16, RoPE'd)
  _Float16* Ks = smem + KS_OFF;   // [KT][ROWH]  key rows (f16, RoPE'd)
  _Float16* Vt = smem + VT_OFF;   // [VDIM][ROWH] V transposed: Vt[vd][key]

  const int qt   = blockIdx.x;          // q tile index within chunk (0..7)
  const int h    = blockIdx.y;          // head
  const int bc   = blockIdx.z;          // batch*chunk block (0..15)
  const int tid  = threadIdx.x;
  const int lane = tid & 31;
  const int w    = tid >> 5;            // wave id, owns q rows w*16..w*16+15
  const int col  = lane & 15;           // C-layout column (q index within 16)
  const bool hihalf = (lane >= 16);
  const int sel  = hihalf ? 8 : 0;      // A-fragment interleave select

  // ---------------- stage Q tile (RoPE + f32->f16) ----------------
  {
    const int r = tid >> 1, hsel = tid & 1;       // 2 threads per row, 64 floats each
    const int pos = qt * QT + r;                  // position within chunk
    const float* src = xq + (((size_t)(bc * CHUNK + pos)) * NHEADS + h) * ZDIM + hsel * 64;
    const float* cr = cs + pos * 64 + hsel * 32;
    const float* sr = sn + pos * 64 + hsel * 32;
    __align__(16) _Float16 tmp[64];
    #pragma unroll
    for (int i = 0; i < 16; ++i) {
      v4f x = *(const v4f*)(src + 4 * i);
      float c0 = cr[2*i], s0 = sr[2*i], c1 = cr[2*i+1], s1 = sr[2*i+1];
      tmp[4*i+0] = (_Float16)(x.x * c0 - x.y * s0);
      tmp[4*i+1] = (_Float16)(x.x * s0 + x.y * c0);
      tmp[4*i+2] = (_Float16)(x.z * c1 - x.w * s1);
      tmp[4*i+3] = (_Float16)(x.z * s1 + x.w * c1);
    }
    v8h* dst = (v8h*)(Qs + r * ROWH + hsel * 64);
    #pragma unroll
    for (int j = 0; j < 8; ++j) dst[j] = *(const v8h*)(tmp + 8 * j);
  }
  __syncthreads();

  // ---------------- persistent Q B-fragments (interleaved K split) ----------------
  // B layout: lanes 0-15 hold K=0..7,16..23, lanes 16-31 hold K=8..15,24..31
  HF qf[4];
  {
    const _Float16* qrow = Qs + (w * 16 + col) * ROWH;
    #pragma unroll
    for (int s = 0; s < 4; ++s) {
      const _Float16* p = qrow + s * 32 + (hihalf ? 8 : 0);
      qf[s].h[0] = *(const v8h*)(p);
      qf[s].h[1] = *(const v8h*)(p + 16);
    }
  }

  // ---------------- flash accumulators (O^T: 16 vd-blocks x 8 rows) ----------------
  const v8f vzero = {0.f,0.f,0.f,0.f,0.f,0.f,0.f,0.f};
  v8f O[16];
  #pragma unroll
  for (int i = 0; i < 16; ++i) O[i] = vzero;
  float m_run = -INFINITY, l_run = 0.f;

  for (int kt = 0; kt <= qt; ++kt) {
    // -------- stage K tile (RoPE) --------
    {
      const int r = tid >> 1, hsel = tid & 1;
      const int pos = kt * KT + r;
      const float* src = xk + (((size_t)(bc * CHUNK + pos)) * NHEADS + h) * ZDIM + hsel * 64;
      const float* cr = cs + pos * 64 + hsel * 32;
      const float* sr = sn + pos * 64 + hsel * 32;
      __align__(16) _Float16 tmp[64];
      #pragma unroll
      for (int i = 0; i < 16; ++i) {
        v4f x = *(const v4f*)(src + 4 * i);
        float c0 = cr[2*i], s0 = sr[2*i], c1 = cr[2*i+1], s1 = sr[2*i+1];
        tmp[4*i+0] = (_Float16)(x.x * c0 - x.y * s0);
        tmp[4*i+1] = (_Float16)(x.x * s0 + x.y * c0);
        tmp[4*i+2] = (_Float16)(x.z * c1 - x.w * s1);
        tmp[4*i+3] = (_Float16)(x.z * s1 + x.w * c1);
      }
      v8h* dst = (v8h*)(Ks + r * ROWH + hsel * 64);
      #pragma unroll
      for (int j = 0; j < 8; ++j) dst[j] = *(const v8h*)(tmp + 8 * j);
    }
    // -------- stage V tile transposed: Vt[vd][key], 4 key rows packed per b64 store ----
    {
      const int g   = tid >> 3;          // 32 row groups: keys 4g..4g+3
      const int sub = tid & 7;           // vd chunk: 32 vds starting sub*32
      const int r0  = g * 4;
      const size_t rstride = (size_t)NHEADS * VDIM;   // floats between key rows
      const float* src = xv + (((size_t)(bc * CHUNK + kt * KT + r0)) * NHEADS + h) * VDIM
                            + sub * 32;
      #pragma unroll
      for (int i = 0; i < 8; ++i) {      // 8 chunks of 4 vd
        v4f x0 = *(const v4f*)(src + 0 * rstride + 4 * i);
        v4f x1 = *(const v4f*)(src + 1 * rstride + 4 * i);
        v4f x2 = *(const v4f*)(src + 2 * rstride + 4 * i);
        v4f x3 = *(const v4f*)(src + 3 * rstride + 4 * i);
        const int vd = sub * 32 + 4 * i;
        v4h p0 = { (_Float16)x0.x, (_Float16)x1.x, (_Float16)x2.x, (_Float16)x3.x };
        v4h p1 = { (_Float16)x0.y, (_Float16)x1.y, (_Float16)x2.y, (_Float16)x3.y };
        v4h p2 = { (_Float16)x0.z, (_Float16)x1.z, (_Float16)x2.z, (_Float16)x3.z };
        v4h p3 = { (_Float16)x0.w, (_Float16)x1.w, (_Float16)x2.w, (_Float16)x3.w };
        *(v4h*)(Vt + (vd + 0) * ROWH + r0) = p0;   // ds_store_b64 (8B aligned: r0 % 4 == 0)
        *(v4h*)(Vt + (vd + 1) * ROWH + r0) = p1;
        *(v4h*)(Vt + (vd + 2) * ROWH + r0) = p2;
        *(v4h*)(Vt + (vd + 3) * ROWH + r0) = p3;
      }
    }
    __syncthreads();

    // -------- prefetch next tile toward L2 while this tile computes --------
    if (kt < qt) {
      const int r = tid >> 1, hsel = tid & 1;
      const size_t nrow = (size_t)(bc * CHUNK + (kt + 1) * KT + r) * NHEADS + h;
      __builtin_prefetch(xk + nrow * ZDIM + hsel * 64, 0, 3);
      __builtin_prefetch(xv + nrow * VDIM + hsel * 128, 0, 3);
    }

    const bool diag = (kt == qt);
    const int nmb = diag ? (w + 1) : 8;   // key blocks needed (causal skip)

    // -------- S^T = K * Q^T  (M=key, N=q), 8 blocks of 16 keys --------
    v8f Sb[8];
    #pragma unroll
    for (int mb = 0; mb < 8; ++mb) {
      v8f acc = vzero;
      if (mb < nmb) {
        const _Float16* krow = Ks + (mb * 16 + col) * ROWH;   // A row M = key
        HF kf[4];   // preload all 4 fragments -> 8 ds_load_b128 in flight, batched waits
        #pragma unroll
        for (int s = 0; s < 4; ++s) {
          kf[s].h[0] = *(const v8h*)(krow + s * 32 + sel);
          kf[s].h[1] = *(const v8h*)(krow + s * 32 + 16 + sel);
        }
        #pragma unroll
        for (int s = 0; s < 4; ++s)
          acc = __builtin_amdgcn_wmma_f32_16x16x32_f16(
              false, kf[s].v, false, qf[s].v, (short)0, acc, false, false);
        if (diag && mb == nmb - 1) {        // only block mb==w touches the diagonal
          #pragma unroll
          for (int r = 0; r < 8; ++r) {
            const int key = mb * 16 + r + sel;      // C layout: M = r + 8*hi
            const int q   = w * 16 + col;
            acc[r] += (key > q) ? NEGB : 0.f;
          }
        }
      } else {
        #pragma unroll
        for (int r = 0; r < 8; ++r) acc[r] = NEGB; // fully masked -> exp()=0
      }
      Sb[mb] = acc;
    }

    // -------- online softmax (per q column; lane pair L / L+16 share a column) --------
    float mt = -INFINITY;
    #pragma unroll
    for (int mb = 0; mb < 8; ++mb)
      #pragma unroll
      for (int r = 0; r < 8; ++r) mt = fmaxf(mt, Sb[mb][r]);
    mt = fmaxf(mt, __shfl_xor(mt, 16, 32));
    const float m_new = fmaxf(m_run, mt);
    const float scale = __expf(m_run - m_new);
    #pragma unroll
    for (int i = 0; i < 16; ++i) O[i] *= scale;

    float rs = 0.f;
    PU pb[8];
    #pragma unroll
    for (int mb = 0; mb < 8; ++mb) {
      #pragma unroll
      for (int j = 0; j < 8; ++j) {
        const float e = __expf(Sb[mb][j] - m_new);
        rs += e;
        pb[mb].h[j] = (_Float16)e;
      }
    }
    rs += __shfl_xor(rs, 16, 32);
    l_run = l_run * scale + rs;
    m_run = m_new;

    // -------- build P^T B-fragments: half-swap between lane pairs --------
    // lo lane needs keys 8..15 of block 2s (held by partner hi lane);
    // hi lane needs keys 0..7 of block 2s+1 (held by partner lo lane).
    v16h P16[4];
    #pragma unroll
    for (int s = 0; s < 4; ++s) {
      PU send = hihalf ? pb[2*s] : pb[2*s+1];
      PU recv;
      #pragma unroll
      for (int k = 0; k < 4; ++k) recv.i[k] = __shfl_xor(send.i[k], 16, 32);
      HF p;
      p.h[0] = hihalf ? recv.h : pb[2*s].h;
      p.h[1] = hihalf ? pb[2*s+1].h : recv.h;
      P16[s] = p.v;
    }

    // -------- O^T += V^T * P^T  (M=vd, N=q) --------
    #pragma unroll
    for (int vb = 0; vb < 16; ++vb) {
      const _Float16* vrow = Vt + (vb * 16 + col) * ROWH;   // A row M = vd
      HF vf[4];   // preload all 4 fragments before the WMMA chain
      #pragma unroll
      for (int s = 0; s < 4; ++s) {
        vf[s].h[0] = *(const v8h*)(vrow + s * 32 + (hihalf ? 16 : 0));
        vf[s].h[1] = *(const v8h*)(vrow + s * 32 + (hihalf ? 16 : 0) + 8);
      }
      v8f acc = O[vb];
      #pragma unroll
      for (int s = 0; s < 4; ++s)
        acc = __builtin_amdgcn_wmma_f32_16x16x32_f16(
            false, vf[s].v, false, P16[s], (short)0, acc, false, false);
      O[vb] = acc;
    }
    __syncthreads();   // LDS reads done before next tile overwrites
  }

  // ---------------- epilogue: normalize + store (lane writes 8 consecutive vd) ------
  const float inv = 1.f / l_run;
  const int qrow = bc * CHUNK + qt * QT + w * 16 + col;
  float* obase = out + ((size_t)qrow * NHEADS + h) * VDIM;
  #pragma unroll
  for (int vb = 0; vb < 16; ++vb) {
    v4f a = { O[vb][0] * inv, O[vb][1] * inv, O[vb][2] * inv, O[vb][3] * inv };
    v4f b = { O[vb][4] * inv, O[vb][5] * inv, O[vb][6] * inv, O[vb][7] * inv };
    *(v4f*)(obase + vb * 16 + sel)     = a;   // vd = vb*16 + 8*hi + r
    *(v4f*)(obase + vb * 16 + sel + 4) = b;
  }
}

extern "C" void kernel_launch(void* const* d_in, const int* in_sizes, int n_in,
                              void* d_out, int out_size, void* d_ws, size_t ws_size,
                              hipStream_t stream) {
  const float* xq = (const float*)d_in[0];
  const float* xk = (const float*)d_in[1];
  const float* xv = (const float*)d_in[2];
  // d_in[3] is the causal mask -- computed analytically in-kernel, unused.
  const float* cs = (const float*)d_in[4];
  const float* sn = (const float*)d_in[5];
  float* out = (float*)d_out;

  static_assert(SMEM_BYTES <= 160 * 1024, "LDS budget (2 WGs per 320KB WGP)");
  hipFuncSetAttribute(reinterpret_cast<const void*>(fa_chunked_rope_kernel),
                      hipFuncAttributeMaxDynamicSharedMemorySize, SMEM_BYTES);

  dim3 grid(CHUNK / QT, NHEADS, 16);   // 8 q-tiles x 8 heads x (BS*SLEN/CHUNK)=16
  fa_chunked_rope_kernel<<<grid, dim3(256), SMEM_BYTES, stream>>>(xq, xk, xv, cs, sn, out);
}
